// TENFastLayer_7017976561855
// MI455X (gfx1250) — matrix operand order repaired and checked by reading, not compiled, not run on hardware
//
#include <hip/hip_runtime.h>
#include <hip/hip_bf16.h>
#include <math.h>

typedef __bf16 bf16_t;
typedef __attribute__((ext_vector_type(16))) __bf16 v16bf;
typedef __attribute__((ext_vector_type(8)))  __bf16 v8bf;
typedef __attribute__((ext_vector_type(8)))  float  v8f;

typedef unsigned int tdm_u32x4 __attribute__((ext_vector_type(4)));
typedef int          tdm_i32x8 __attribute__((ext_vector_type(8)));
typedef int          tdm_i32x4 __attribute__((ext_vector_type(4)));

#define D_MODEL 1024
#define KEIG    64
#define TWOK    128
#define MLP_DIM 4096
#define BSZ     4
#define TSEQ    4096
#define MTOK    (BSZ*TSEQ)   // 16384 tokens
#define NPROJ   1280         // 128 beta + 128 delta + 1024 gate cols

// LDS tile geometry: 128 rows x 32 bf16 (64B) + 16B pad -> 80B row stride
#define TILE_ROWS   128
#define TILE_K      32
#define ROW_ELEMS   40                    // 32 data + 8 pad bf16
#define TILE_ELEMS  (TILE_ROWS*ROW_ELEMS) // 5120
#define TILE_BYTES  (TILE_ELEMS*2)        // 10240

__device__ __forceinline__ float sigmf(float x) { return 1.0f / (1.0f + __expf(-x)); }

// ---------------------------------------------------------------- TDM issue
// One Tensor-Data-Mover descriptor: copy a 128-row x 32-elem (bf16) tile,
// row padding 16B so LDS row stride is 80B (bank-conflict-free ds_load_b128).
__device__ __forceinline__ void tdm_load_tile_128x32(const void* gsrc, unsigned lds_byte_off,
                                                     unsigned row_len, unsigned nrows) {
  unsigned long long ga = (unsigned long long)(uintptr_t)gsrc;
  tdm_u32x4 g0;
  g0[0] = 1u;                                   // count=1, user descriptor
  g0[1] = lds_byte_off;                         // lds_addr [63:32]
  g0[2] = (unsigned)ga;                         // global_addr[31:0]
  g0[3] = ((unsigned)(ga >> 32) & 0x01FFFFFFu)  // global_addr[56:32]
          | (2u << 30);                         // type = 2 ("image")
  tdm_i32x8 g1;
  g1[0] = (int)((1u << 16)        // data_size = 2 bytes
              | (1u << 20)        // pad_enable
              | (3u << 22)        // pad_interval = 16 dwords (one 64B row)
              | (3u << 25));      // pad_amount   = 4 dwords (16B)
  g1[1] = (int)((row_len & 0xFFFFu) << 16);                     // tensor_dim0[15:0]
  g1[2] = (int)((row_len >> 16) | ((nrows & 0xFFFFu) << 16));   // dim0[31:16] | dim1[15:0]
  g1[3] = (int)((nrows >> 16) | ((unsigned)TILE_K << 16));      // dim1[31:16] | tile_dim0
  g1[4] = (int)(unsigned)TILE_ROWS;                             // tile_dim1 | tile_dim2=0
  g1[5] = (int)row_len;                                         // tensor_dim0_stride[31:0]
  g1[6] = 0;                                                    // stride0[47:32] | stride1[15:0]
  g1[7] = 0;
  tdm_i32x4 z4 = {0, 0, 0, 0};
#if defined(__clang_major__) && (__clang_major__ >= 23)
  tdm_i32x8 z8 = {0, 0, 0, 0, 0, 0, 0, 0};
  __builtin_amdgcn_tensor_load_to_lds(g0, g1, z4, z4, z8, 0);
#else
  __builtin_amdgcn_tensor_load_to_lds(g0, g1, z4, z4, 0);
#endif
}

// ------------------------------------------------------------- pack fp32->bf16
__global__ void pack_bf16_kernel(const float* __restrict__ src, bf16_t* __restrict__ dst, int n) {
  int i = blockIdx.x * blockDim.x + threadIdx.x;
  if (i < n) dst[i] = (bf16_t)src[i];
}

// ---------------------------------------- fold R rotation into W_out (bf16 out)
__global__ void fold_wout_kernel(const float* __restrict__ W_out, const float* __restrict__ R,
                                 bf16_t* __restrict__ wf) {
  int t = blockIdx.x * blockDim.x + threadIdx.x;
  if (t >= D_MODEL * TWOK) return;
  int d = t >> 7;
  int c = t & 127;
  int off = (c < KEIG) ? 0 : KEIG;
  int cc = c & 63;
  int h = cc >> 4;
  int kc = cc & 15;
  const float* wrow = W_out + (size_t)d * TWOK + off + h * 16;
  const float* rh = R + h * 256 + kc;
  float acc = 0.f;
#pragma unroll
  for (int j = 0; j < 16; ++j) acc += wrow[j] * rh[j * 16];
  wf[(size_t)d * TWOK + c] = (bf16_t)acc;
}

// ------------------------------------------------------ LayerNorm -> bf16 rows
__global__ __launch_bounds__(256) void ln_bf16_kernel(const float* __restrict__ x,
                                                      const float* __restrict__ g,
                                                      const float* __restrict__ be,
                                                      bf16_t* __restrict__ out) {
  __shared__ float red[256];
  int row = blockIdx.x, tid = threadIdx.x;
  const float* xr = x + (size_t)row * D_MODEL;
  float4 v = *(const float4*)(xr + tid * 4);
  red[tid] = v.x + v.y + v.z + v.w;
  __syncthreads();
  for (int o = 128; o > 0; o >>= 1) { if (tid < o) red[tid] += red[tid + o]; __syncthreads(); }
  float mean = red[0] * (1.0f / D_MODEL);
  __syncthreads();
  float d0 = v.x - mean, d1 = v.y - mean, d2 = v.z - mean, d3 = v.w - mean;
  red[tid] = d0 * d0 + d1 * d1 + d2 * d2 + d3 * d3;
  __syncthreads();
  for (int o = 128; o > 0; o >>= 1) { if (tid < o) red[tid] += red[tid + o]; __syncthreads(); }
  float rstd = rsqrtf(red[0] * (1.0f / D_MODEL) + 1e-5f);
  int c = tid * 4;
  bf16_t* orow = out + (size_t)row * D_MODEL + c;
  orow[0] = (bf16_t)(d0 * rstd * g[c + 0] + be[c + 0]);
  orow[1] = (bf16_t)(d1 * rstd * g[c + 1] + be[c + 1]);
  orow[2] = (bf16_t)(d2 * rstd * g[c + 2] + be[c + 2]);
  orow[3] = (bf16_t)(d3 * rstd * g[c + 3] + be[c + 3]);
}

// --------------------------------------------------------- WMMA bf16 GEMM core
// C[m,n] = sum_k A[m,k]*W[n,k]; A:[M,Kdim] bf16, W:[N,Kdim] bf16, both row-major.
// Block tile 128x128, TDM double-buffered LDS staging, 8 waves of 32Mx64N each.
template <class Epi>
__global__ __launch_bounds__(256) void gemm_wmma_kernel(const bf16_t* __restrict__ A,
                                                        const bf16_t* __restrict__ W,
                                                        int Kdim, int N, Epi epi) {
  // [buf][0]=A tile, [buf][1]=B tile; padded rows (80B) for conflict-free b128.
  // NOTE: single shared object in this kernel -> LDS offset 0 assumed for TDM.
  __shared__ __align__(128) bf16_t lds_tiles[2][2][TILE_ELEMS];

  const int tid = threadIdx.x;
  const int lane = tid & 31;
  const int wv = tid >> 5;
  const int wm = wv & 3, wn = wv >> 2;
  const int lm = lane & 15;
  const int hi = lane >> 4;
  const int m0 = blockIdx.y * 128 + wm * 32;
  const int n0 = blockIdx.x * 128 + wn * 64;

  // give the LDS object a visible store so TDM-written data isn't folded away
  {
    v8bf z;
#pragma unroll
    for (int e = 0; e < 8; ++e) z[e] = (bf16_t)0.0f;
    v8bf* p = (v8bf*)&lds_tiles[0][0][0];
    for (int i = tid; i < (int)(2 * 2 * TILE_ELEMS / 8); i += 256) p[i] = z;
  }
  __syncthreads();

  const bf16_t* gA = A + (size_t)(blockIdx.y * 128) * Kdim;
  const bf16_t* gB = W + (size_t)(blockIdx.x * 128) * Kdim;

  if (wv == 0) {
    tdm_load_tile_128x32(gA, 0 * TILE_BYTES, (unsigned)Kdim, (unsigned)MTOK);
    tdm_load_tile_128x32(gB, 1 * TILE_BYTES, (unsigned)Kdim, (unsigned)N);
    __builtin_amdgcn_s_wait_tensorcnt(0);
  }
  __syncthreads();

  v8f acc[2][4];
  v8f zero = {0.f, 0.f, 0.f, 0.f, 0.f, 0.f, 0.f, 0.f};
#pragma unroll
  for (int s = 0; s < 2; ++s)
#pragma unroll
    for (int j = 0; j < 4; ++j) acc[s][j] = zero;

  const int nk = Kdim / TILE_K;
  for (int ks = 0; ks < nk; ++ks) {
    const int buf = ks & 1;
    if (wv == 0 && ks + 1 < nk) {  // prefetch next k-tile into other buffer
      int k0 = (ks + 1) * TILE_K;
      tdm_load_tile_128x32(gA + k0, (unsigned)((buf ^ 1) * 2 + 0) * TILE_BYTES,
                           (unsigned)Kdim, (unsigned)MTOK);
      tdm_load_tile_128x32(gB + k0, (unsigned)((buf ^ 1) * 2 + 1) * TILE_BYTES,
                           (unsigned)Kdim, (unsigned)N);
    }

    // A fragments: 2 sub-tiles of 16 rows (split-K-halves layout)
    v16bf af[2];
#pragma unroll
    for (int s = 0; s < 2; ++s) {
      const bf16_t* pa = &lds_tiles[buf][0][(wm * 32 + s * 16 + lm) * ROW_ELEMS + hi * 8];
      v8bf a_lo = *(const v8bf*)pa;
      v8bf a_hi = *(const v8bf*)(pa + 16);
      af[s] = __builtin_shufflevector(a_lo, a_hi,
          0, 1, 2, 3, 4, 5, 6, 7, 8, 9, 10, 11, 12, 13, 14, 15);
    }
    // B fragments reused across both A sub-tiles
#pragma unroll
    for (int j = 0; j < 4; ++j) {
      const bf16_t* pb = &lds_tiles[buf][1][(wn * 64 + j * 16 + lm) * ROW_ELEMS + hi * 8];
      v8bf b_lo = *(const v8bf*)pb;
      v8bf b_hi = *(const v8bf*)(pb + 16);
      v16bf bfrag = __builtin_shufflevector(b_lo, b_hi,
          0, 1, 2, 3, 4, 5, 6, 7, 8, 9, 10, 11, 12, 13, 14, 15);
      acc[0][j] = __builtin_amdgcn_wmma_f32_16x16x32_bf16(
          false, af[0], false, bfrag, (short)0, acc[0][j], false, false);
      acc[1][j] = __builtin_amdgcn_wmma_f32_16x16x32_bf16(
          false, af[1], false, bfrag, (short)0, acc[1][j], false, false);
    }

    if (wv == 0 && ks + 1 < nk) __builtin_amdgcn_s_wait_tensorcnt(0);
    __syncthreads();
  }

#pragma unroll
  for (int s = 0; s < 2; ++s)
#pragma unroll
    for (int j = 0; j < 4; ++j)
#pragma unroll
      for (int i = 0; i < 8; ++i)
        epi(m0 + s * 16 + i + hi * 8, n0 + j * 16 + lm, acc[s][j][i]);
}

// ----------------------------------------------------------------- epilogues
struct EpiProj {  // beta/delta -> fp32 proj (+b_sel); gate cols -> sigmoid bf16
  float* proj; bf16_t* gate; const float* b_sel; const float* b_g;
  __device__ void operator()(int m, int n, float v) const {
    if (n < 256) {
      if (n >= 128) v += b_sel[n - 128];
      proj[(size_t)m * 256 + n] = v;
    } else {
      int d = n - 256;
      gate[(size_t)m * D_MODEL + d] = (bf16_t)sigmf(v + b_g[d]);
    }
  }
};
struct EpiMid {  // mid = x + gate * h
  float* mid; const float* x; const bf16_t* gate;
  __device__ void operator()(int m, int n, float v) const {
    size_t idx = (size_t)m * D_MODEL + n;
    mid[idx] = x[idx] + (float)gate[idx] * v;
  }
};
struct EpiSilu {  // u = silu(v + b1) -> bf16
  bf16_t* u; const float* b1;
  __device__ void operator()(int m, int n, float v) const {
    float t = v + b1[n];
    u[(size_t)m * MLP_DIM + n] = (bf16_t)(t * sigmf(t));
  }
};
struct EpiOut {  // out = mid + v + b2
  float* out; const float* mid; const float* b2;
  __device__ void operator()(int m, int n, float v) const {
    size_t idx = (size_t)m * D_MODEL + n;
    out[idx] = mid[idx] + v + b2[n];
  }
};

// ------------------------------------------- complex prefix scan over T=4096
__global__ __launch_bounds__(256) void scan_kernel(const float* __restrict__ proj,
                                                   const float* __restrict__ log_decay,
                                                   const float* __restrict__ frequency,
                                                   bf16_t* __restrict__ cvec) {
  __shared__ float sAr[256], sAi[256], sBr[256], sBi[256];
  const int b = blockIdx.x >> 6;
  const int k = blockIdx.x & 63;
  const int tid = threadIdx.x;
  const float ld = log_decay[k];
  const float fq = frequency[k];
  const size_t rbase = ((size_t)b * TSEQ + (size_t)tid * 16) * 256;

  float Ar = 1.f, Ai = 0.f, Br = 0.f, Bi = 0.f;
  const float* p = proj + rbase;
  for (int i = 0; i < 16; ++i) {
    float br = p[k], bi = p[64 + k];
    float mag = sigmf(ld + 0.1f * tanhf(p[128 + k]));
    float f = fq + 0.05f * tanhf(p[192 + k]);
    float ar = mag * __cosf(f), ai = mag * __sinf(f);
    float nAr = ar * Ar - ai * Ai;
    float nAi = ar * Ai + ai * Ar;
    float nBr = ar * Br - ai * Bi + br;
    float nBi = ar * Bi + ai * Br + bi;
    Ar = nAr; Ai = nAi; Br = nBr; Bi = nBi;
    p += 256;
  }
  sAr[tid] = Ar; sAi[tid] = Ai; sBr[tid] = Br; sBi[tid] = Bi;
  __syncthreads();

  for (int o = 1; o < 256; o <<= 1) {
    float pAr = 0, pAi = 0, pBr = 0, pBi = 0;
    bool act = (tid >= o);
    if (act) { pAr = sAr[tid - o]; pAi = sAi[tid - o]; pBr = sBr[tid - o]; pBi = sBi[tid - o]; }
    float cAr = sAr[tid], cAi = sAi[tid], cBr = sBr[tid], cBi = sBi[tid];
    __syncthreads();
    if (act) {
      sAr[tid] = cAr * pAr - cAi * pAi;
      sAi[tid] = cAr * pAi + cAi * pAr;
      sBr[tid] = cAr * pBr - cAi * pBi + cBr;
      sBi[tid] = cAr * pBi + cAi * pBr + cBi;
    }
    __syncthreads();
  }
  float cr = (tid == 0) ? 0.f : sBr[tid - 1];
  float ci = (tid == 0) ? 0.f : sBi[tid - 1];

  p = proj + rbase;
  bf16_t* op = cvec + ((size_t)b * TSEQ + (size_t)tid * 16) * TWOK;
  for (int i = 0; i < 16; ++i) {
    float br = p[k], bi = p[64 + k];
    float mag = sigmf(ld + 0.1f * tanhf(p[128 + k]));
    float f = fq + 0.05f * tanhf(p[192 + k]);
    float ar = mag * __cosf(f), ai = mag * __sinf(f);
    float ncr = ar * cr - ai * ci + br;
    float nci = ar * ci + ai * cr + bi;
    cr = ncr; ci = nci;
    op[k] = (bf16_t)cr;
    op[64 + k] = (bf16_t)ci;
    p += 256;
    op += TWOK;
  }
}

// ------------------------------------------------------------------- launcher
extern "C" void kernel_launch(void* const* d_in, const int* in_sizes, int n_in,
                              void* d_out, int out_size, void* d_ws, size_t ws_size,
                              hipStream_t stream) {
  (void)in_sizes; (void)n_in; (void)out_size; (void)ws_size;
  const float* x         = (const float*)d_in[0];
  const float* W_in      = (const float*)d_in[1];
  const float* log_decay = (const float*)d_in[2];
  const float* frequency = (const float*)d_in[3];
  const float* W_sel     = (const float*)d_in[4];
  const float* b_sel     = (const float*)d_in[5];
  const float* R         = (const float*)d_in[6];
  const float* W_out     = (const float*)d_in[7];
  const float* W_g       = (const float*)d_in[8];
  const float* b_g       = (const float*)d_in[9];
  const float* g1        = (const float*)d_in[10];
  const float* be1       = (const float*)d_in[11];
  const float* g2        = (const float*)d_in[12];
  const float* be2       = (const float*)d_in[13];
  const float* W1        = (const float*)d_in[14];
  const float* b1        = (const float*)d_in[15];
  const float* W2        = (const float*)d_in[16];
  const float* b2        = (const float*)d_in[17];
  float* out = (float*)d_out;

  char* ws = (char*)d_ws;
  size_t off = 0;
  auto take = [&](size_t bytes) -> char* {
    char* p = ws + off;
    off = (off + bytes + 255) & ~(size_t)255;
    return p;
  };
  bf16_t* xn   = (bf16_t*)take((size_t)MTOK * D_MODEL * 2);   // x_norm, reused as x2
  bf16_t* wcat = (bf16_t*)take((size_t)NPROJ * D_MODEL * 2);  // [W_in;W_sel;W_g] bf16
  float*  proj = (float*) take((size_t)MTOK * 256 * 4);       // beta/delta fp32
  bf16_t* gate = (bf16_t*)take((size_t)MTOK * D_MODEL * 2);   // sigmoid gate bf16
  bf16_t* cvec = (bf16_t*)take((size_t)MTOK * TWOK * 2);      // scan output [c_r|c_i]
  bf16_t* wf   = (bf16_t*)take((size_t)D_MODEL * TWOK * 2);   // R-folded W_out bf16
  float*  mid  = (float*) take((size_t)MTOK * D_MODEL * 4);   // post-attention residual
  bf16_t* w1b  = (bf16_t*)take((size_t)MLP_DIM * D_MODEL * 2);
  bf16_t* w2b  = (bf16_t*)take((size_t)D_MODEL * MLP_DIM * 2);
  bf16_t* ubuf = (bf16_t*)take((size_t)MTOK * MLP_DIM * 2);   // silu activations

  auto cdiv = [](int a, int b) { return (a + b - 1) / b; };

  pack_bf16_kernel<<<cdiv(TWOK * D_MODEL, 256), 256, 0, stream>>>(W_in, wcat, TWOK * D_MODEL);
  pack_bf16_kernel<<<cdiv(TWOK * D_MODEL, 256), 256, 0, stream>>>(W_sel, wcat + (size_t)TWOK * D_MODEL, TWOK * D_MODEL);
  pack_bf16_kernel<<<cdiv(D_MODEL * D_MODEL, 256), 256, 0, stream>>>(W_g, wcat + (size_t)256 * D_MODEL, D_MODEL * D_MODEL);
  pack_bf16_kernel<<<cdiv(MLP_DIM * D_MODEL, 256), 256, 0, stream>>>(W1, w1b, MLP_DIM * D_MODEL);
  pack_bf16_kernel<<<cdiv(D_MODEL * MLP_DIM, 256), 256, 0, stream>>>(W2, w2b, D_MODEL * MLP_DIM);
  fold_wout_kernel<<<cdiv(D_MODEL * TWOK, 256), 256, 0, stream>>>(W_out, R, wf);

  ln_bf16_kernel<<<MTOK, 256, 0, stream>>>(x, g1, be1, xn);

  // GEMM1: [16384x1024] x [1280x1024]^T -> proj + gate
  gemm_wmma_kernel<<<dim3(NPROJ / 128, MTOK / 128), 256, 0, stream>>>(
      xn, wcat, D_MODEL, NPROJ, EpiProj{proj, gate, b_sel, b_g});

  scan_kernel<<<BSZ * KEIG, 256, 0, stream>>>(proj, log_decay, frequency, cvec);

  // GEMM2: [16384x128] x [1024x128]^T + gated residual -> mid
  gemm_wmma_kernel<<<dim3(D_MODEL / 128, MTOK / 128), 256, 0, stream>>>(
      cvec, wf, TWOK, D_MODEL, EpiMid{mid, x, gate});

  ln_bf16_kernel<<<MTOK, 256, 0, stream>>>(mid, g2, be2, xn);

  // GEMM3: [16384x1024] x [4096x1024]^T, silu -> ubuf
  gemm_wmma_kernel<<<dim3(MLP_DIM / 128, MTOK / 128), 256, 0, stream>>>(
      xn, w1b, D_MODEL, MLP_DIM, EpiSilu{ubuf, b1});

  // GEMM4: [16384x4096] x [1024x4096]^T + mid + b2 -> out
  gemm_wmma_kernel<<<dim3(D_MODEL / 128, MTOK / 128), 256, 0, stream>>>(
      ubuf, w2b, MLP_DIM, D_MODEL, EpiOut{out, mid, b2});
}
